// LSTMAutoreg_19997367730782
// MI455X (gfx1250) — matrix operand order repaired and checked by reading, not compiled, not run on hardware
//
#include <hip/hip_runtime.h>

// ---------------------------------------------------------------------------
// LSTM on MI455X (gfx1250), persistent wave32 WMMA kernel, v4.
//  - Combined, gate-interleaved, transposed bf16 weights WcT[4096][2048]
//      (p = hcol*4 + gate, k<1024 -> Wi, k>=1024 -> Wh)  -> 16MB, L2 resident
//  - A operand (x_t bf16 + h bf16) staged in DOUBLE-BUFFERED LDS k-panels:
//    fill of panel kp+1 issued before compute of panel kp (latency hidden)
//  - depth-1 software pipeline of fragment loads inside the compute loop
//  - gates tile in LDS, c-state in registers, single grid barrier per step
// ---------------------------------------------------------------------------

typedef __bf16 v16bf  __attribute__((ext_vector_type(16)));
typedef __bf16 bf16x8 __attribute__((ext_vector_type(8)));
typedef float  v8f    __attribute__((ext_vector_type(8)));
typedef float  f32x4  __attribute__((ext_vector_type(4)));

#define T_STEPS  512
#define BATCH    64
#define DIM      1024
#define HID      1024
#define KTOT     2048          // DIM + HID
#define G4       4096
#define NWG      128           // persistent grid
#define NTHREADS 128
#define PANEL    256           // K-panel staged in LDS
#define MROWS    32            // batch rows per WG (M-half)
#define APITCH   264           // bf16 per LDS A row (256 + 8 pad)
#define ABUF     (MROWS * APITCH)
#define GPITCH   68            // f32 per LDS gates row (64 + 4 pad)

__device__ __forceinline__ float fast_sigmoid(float x) {
  return 1.0f / (1.0f + __expf(-x));
}
__device__ __forceinline__ float fast_tanh(float x) {
  float e = __expf(2.0f * x);
  return (e - 1.0f) / (e + 1.0f);
}

__device__ __forceinline__ void grid_barrier(int* ctr, int target) {
  __threadfence();
  __syncthreads();
  if (threadIdx.x == 0) {
    __hip_atomic_fetch_add(ctr, 1, __ATOMIC_ACQ_REL, __HIP_MEMORY_SCOPE_AGENT);
    while (__hip_atomic_load(ctr, __ATOMIC_ACQUIRE, __HIP_MEMORY_SCOPE_AGENT) < target) {
      __builtin_amdgcn_s_sleep(2);
    }
  }
  __syncthreads();
  __threadfence();
}

// Stage one K-panel of the A operand (rows [rowbase, rowbase+32)) into LDS.
__device__ __forceinline__ void fill_panel(__bf16* __restrict__ dst, int kp,
                                           const float* __restrict__ xt,
                                           const __bf16* __restrict__ hprev,
                                           int rowbase, int tid) {
  const int kbase = kp * PANEL;
  if (kp < 4) {
    // from x_t (f32 -> bf16 in-register)
    #pragma unroll
    for (int i = 0; i < 8; ++i) {
      int idx = (i * NTHREADS + tid) * 8;           // [0, 8192)
      int rl = idx >> 8;
      int kloc = idx & 255;
      const f32x4* src =
          (const f32x4*)(xt + (size_t)(rowbase + rl) * DIM + kbase + kloc);
      f32x4 s0 = src[0], s1 = src[1];
      bf16x8 v;
      #pragma unroll
      for (int q = 0; q < 4; ++q) { v[q] = (__bf16)s0[q]; v[4 + q] = (__bf16)s1[q]; }
      *(bf16x8*)(&dst[rl * APITCH + kloc]) = v;
    }
  } else {
    // from h_{t-1} (already bf16)
    #pragma unroll
    for (int i = 0; i < 8; ++i) {
      int idx = (i * NTHREADS + tid) * 8;
      int rl = idx >> 8;
      int kloc = idx & 255;
      *(bf16x8*)(&dst[rl * APITCH + kloc]) =
          *(const bf16x8*)(hprev + (size_t)(rowbase + rl) * HID +
                           (kbase - DIM) + kloc);
    }
  }
}

// Build WcT[p][k]: p = permuted gate column (hcol*4+gate), k = Wi|Wh rows, bf16.
__global__ void build_wct_kernel(const float* __restrict__ Wi,
                                 const float* __restrict__ Wh,
                                 __bf16* __restrict__ WcT) {
  size_t gid = (size_t)blockIdx.x * blockDim.x + threadIdx.x;
  if (gid >= (size_t)G4 * KTOT) return;
  int k = (int)(gid & (KTOT - 1));
  int p = (int)(gid >> 11);
  int col = (p & 3) * HID + (p >> 2);            // original gate-major column
  float v = (k < DIM) ? Wi[(size_t)k * G4 + col]
                      : Wh[(size_t)(k - DIM) * G4 + col];
  WcT[gid] = (__bf16)v;
}

__global__ void init_state_kernel(const float* __restrict__ h0,
                                  __bf16* __restrict__ hbuf0,
                                  int* __restrict__ ctr) {
  int gid = blockIdx.x * blockDim.x + threadIdx.x;
  if (gid == 0) *ctr = 0;
  if (gid < BATCH * HID) hbuf0[gid] = (__bf16)h0[gid];
}

__global__ void __launch_bounds__(NTHREADS, 1)
lstm_persistent(const float* __restrict__ x,
                const float* __restrict__ c0,
                const float* __restrict__ bias,
                float* __restrict__ out,
                const __bf16* __restrict__ WcT,   // [4096][2048] bf16
                __bf16* __restrict__ hbuf0,       // [64][1024] bf16 ping
                __bf16* __restrict__ hbuf1,       // [64][1024] bf16 pong
                int* __restrict__ barrier_ctr) {
  __shared__ __bf16 Alds[2 * ABUF];         // 33.8 KB: double-buffered A panel
  __shared__ float  Glds[MROWS * GPITCH];   // 8.7 KB: gate tile

  const int tid  = threadIdx.x;
  const int lane = tid & 31;
  const int wave = tid >> 5;             // wave v owns N-tile v (16 p-cols)
  const int wg   = blockIdx.x;
  const int nwg  = gridDim.x;
  const int r16  = lane & 15;
  const int hs   = lane >> 4;            // fragment half-select (A/B k offset 16*hs, C/D row offset 8*hs)

  const int hgrp  = wg >> 1;             // 0..63 : block of 16 h-columns
  const int mhalf = wg & 1;              // 0..1  : rows [mhalf*32, +32)
  const int rowbase = mhalf * MROWS;

  // B fragment column (permuted) for this lane
  const int pcol = hgrp * 64 + wave * 16 + r16;
  const __bf16* __restrict__ bcol = WcT + (size_t)pcol * KTOT;
  const float biasv = bias[(pcol & 3) * HID + (pcol >> 2)];

  // c-state in registers: e = tid + j*128 -> (row e>>4, hcol hgrp*16+(e&15))
  float creg[4];
  #pragma unroll
  for (int j = 0; j < 4; ++j) {
    int e = tid + j * NTHREADS;
    int grow = rowbase + (e >> 4);
    int hc = hgrp * 16 + (e & 15);
    creg[j] = c0[grow * HID + hc];
  }

  int barNum = 0;
  for (int t = 0; t < T_STEPS; ++t) {
    const __bf16* __restrict__ hprev = (t & 1) ? hbuf1 : hbuf0;
    __bf16* __restrict__ hnext = (t & 1) ? hbuf0 : hbuf1;
    const float* __restrict__ xt = x + (size_t)t * BATCH * DIM;

    v8f acc[2];
    #pragma unroll
    for (int m = 0; m < 2; ++m) acc[m] = (v8f){0, 0, 0, 0, 0, 0, 0, 0};

    // prologue: stage panel 0 into buffer 0
    fill_panel(Alds, 0, xt, hprev, rowbase, tid);
    __syncthreads();

    for (int kp = 0; kp < 8; ++kp) {
      const int kbase = kp * PANEL;
      __bf16* curA = Alds + (kp & 1) * ABUF;
      __bf16* nxtA = Alds + ((kp + 1) & 1) * ABUF;

      // ---- issue fill of NEXT panel into other buffer (overlaps compute) ---
      // Safe: last reader of nxtA finished at the barrier ending iter kp-1.
      if (kp < 7) {
        fill_panel(nxtA, kp + 1, xt, hprev, rowbase, tid);
        __builtin_prefetch(bcol + kbase + PANEL + hs * 16, 0, 1);
      }

      // ---- compute: 8 k-steps, depth-1 software pipeline -------------------
      {
        bf16x8 bc0, bc1, ac0[2], ac1[2];
        // prologue: fragments for ks = 0
        {
          const bf16x8* bp = (const bf16x8*)(bcol + kbase + hs * 16);
          bc0 = bp[0];
          bc1 = bp[1];
          #pragma unroll
          for (int m = 0; m < 2; ++m) {
            const __bf16* ap = &curA[(m * 16 + r16) * APITCH + hs * 16];
            ac0[m] = *(const bf16x8*)ap;
            ac1[m] = *(const bf16x8*)(ap + 8);
          }
        }
        #pragma unroll
        for (int ks = 0; ks < PANEL; ks += 32) {
          bf16x8 bn0, bn1, an0[2], an1[2];
          if (ks + 32 < PANEL) {
            // issue next k-step's loads BEFORE consuming current fragments
            const bf16x8* bp =
                (const bf16x8*)(bcol + kbase + ks + 32 + hs * 16);
            bn0 = bp[0];
            bn1 = bp[1];
            #pragma unroll
            for (int m = 0; m < 2; ++m) {
              const __bf16* ap =
                  &curA[(m * 16 + r16) * APITCH + ks + 32 + hs * 16];
              an0[m] = *(const bf16x8*)ap;
              an1[m] = *(const bf16x8*)(ap + 8);
            }
          }
          v16bf bm;
          #pragma unroll
          for (int q = 0; q < 8; ++q) { bm[q] = bc0[q]; bm[q + 8] = bc1[q]; }
          #pragma unroll
          for (int m = 0; m < 2; ++m) {
            v16bf am;
            #pragma unroll
            for (int q = 0; q < 8; ++q) { am[q] = ac0[m][q]; am[q + 8] = ac1[m][q]; }
            acc[m] = __builtin_amdgcn_wmma_f32_16x16x32_bf16(
                false, am, false, bm, (short)0, acc[m], false, false);
          }
          if (ks + 32 < PANEL) {
            bc0 = bn0;
            bc1 = bn1;
            #pragma unroll
            for (int m = 0; m < 2; ++m) { ac0[m] = an0[m]; ac1[m] = an1[m]; }
          }
        }
      }
      __syncthreads();   // publish next panel's fill; retire current buffer
    }

    // ---- dump gate tiles (+bias) to LDS; C/D layout row = r + hs*8 ---------
    #pragma unroll
    for (int m = 0; m < 2; ++m) {
      #pragma unroll
      for (int r = 0; r < 8; ++r) {
        int rl = m * 16 + r + hs * 8;
        Glds[rl * GPITCH + wave * 16 + r16] = acc[m][r] + biasv;
      }
    }
    __syncthreads();

    // ---- fused gate nonlinearity + cell update (c in registers) ------------
    #pragma unroll
    for (int j = 0; j < 4; ++j) {
      int e = tid + j * NTHREADS;
      int rl = e >> 4;
      int jl = e & 15;
      const float* g = &Glds[rl * GPITCH + jl * 4];  // i,f,g,o adjacent (perm)
      float iv = fast_sigmoid(g[0]);
      float fv = fast_sigmoid(g[1]);
      float gv = fast_tanh(g[2]);
      float ov = fast_sigmoid(g[3]);
      float cn = fv * creg[j] + iv * gv;
      creg[j] = cn;
      float hn = ov * fast_tanh(cn);
      int grow = rowbase + rl;
      int hc = hgrp * 16 + jl;
      hnext[grow * HID + hc] = (__bf16)hn;
      out[(size_t)t * (BATCH * HID) + grow * HID + hc] = hn;
    }

    // ---- single grid barrier per step: publish h_t, protect LDS reuse ------
    ++barNum;
    grid_barrier(barrier_ctr, barNum * nwg);
  }
}

extern "C" void kernel_launch(void* const* d_in, const int* in_sizes, int n_in,
                              void* d_out, int out_size, void* d_ws, size_t ws_size,
                              hipStream_t stream) {
  const float* x  = (const float*)d_in[0];   // [512,64,1024]
  const float* c0 = (const float*)d_in[1];   // [64,1024]
  const float* h0 = (const float*)d_in[2];   // [64,1024]
  const float* Wi = (const float*)d_in[3];   // [1024,4096]
  const float* Wh = (const float*)d_in[4];   // [1024,4096]
  const float* b  = (const float*)d_in[5];   // [4096]
  float* out = (float*)d_out;                // [512,64,1024]

  char* ws = (char*)d_ws;
  const size_t OFF_WCT = 0;                                   // 16 MB bf16
  const size_t OFF_H0  = OFF_WCT + (size_t)G4 * KTOT * 2;
  const size_t OFF_H1  = OFF_H0 + (size_t)BATCH * HID * 2;
  const size_t OFF_CTR = OFF_H1 + (size_t)BATCH * HID * 2;

  __bf16* WcT = (__bf16*)(ws + OFF_WCT);
  __bf16* hb0 = (__bf16*)(ws + OFF_H0);
  __bf16* hb1 = (__bf16*)(ws + OFF_H1);
  int*    ctr = (int*)(ws + OFF_CTR);

  const size_t nW = (size_t)G4 * KTOT;
  build_wct_kernel<<<(int)((nW + 255) / 256), 256, 0, stream>>>(Wi, Wh, WcT);
  init_state_kernel<<<(BATCH * HID + 255) / 256, 256, 0, stream>>>(h0, hb0, ctr);

  lstm_persistent<<<NWG, NTHREADS, 0, stream>>>(x, c0, b, out, WcT, hb0, hb1, ctr);
}
